// EncoderLayer_2241972928842
// MI455X (gfx1250) — matrix operand order repaired and checked by reading, not compiled, not run on hardware
//
#include <hip/hip_runtime.h>
#include <hip/hip_bf16.h>

typedef __bf16 bf16;
typedef __attribute__((ext_vector_type(16))) __bf16 bf16x16;
typedef __attribute__((ext_vector_type(8)))  float  v8f;

#define BDIM 4
#define LSEQ 2048
#define DMOD 512
#define NH   8
#define DK   64
#define FF   2048
#define NTOP 40
#define SK   40

// ---------------------------------------------------------------- cvt f32->bf16
__global__ void cvt_f32_bf16(const float* __restrict__ s, bf16* __restrict__ d, int n) {
    int i = blockIdx.x * 256 + threadIdx.x;
    if (i < n) d[i] = (bf16)s[i];
}

// ---------------------------------------------------------------- WMMA GEMM
// C[M,N] = A[M,K] @ W[N,K]^T + bias, optional exact GELU, f32 and/or bf16 out.
// 256 threads = 8 waves; block tile 128x64; staged K-slab of 64 (2 WMMA k-steps);
// LDS double-buffered, filled with CDNA5 async global->LDS DMA (ASYNCcnt).
#define BM 128
#define BN 64
#define BKS 64   // staged K per LDS buffer
#define LDT 72   // LDS row stride in bf16 (144B: 16B-aligned, conflict-free 36-dword stride)

__device__ __forceinline__ bf16x16 frag_a(const bf16* tile, int lane) {
    // 16-bit A 16x32: lanes 0-15 row M=lane, K=0..7 & 16..23; lanes 16-31 K=8..15 & 24..31
    int half = lane >> 4, r = lane & 15;
    const bf16* p = tile + r * LDT + half * 8;
    bf16x16 v;
#pragma unroll
    for (int i = 0; i < 8; ++i) v[i] = p[i];
#pragma unroll
    for (int i = 0; i < 8; ++i) v[8 + i] = p[16 + i];
    return v;
}
__device__ __forceinline__ bf16x16 frag_b(const bf16* tile, int lane) {
    // 16-bit B 32x16: lanes 0-15 col N=lane K=0..7 & 16..23; lanes 16-31 K=8..15 & 24..31
    int half = lane >> 4, n = lane & 15;
    const bf16* p = tile + n * LDT + half * 8;
    bf16x16 v;
#pragma unroll
    for (int i = 0; i < 8; ++i) v[i] = p[i];
#pragma unroll
    for (int i = 0; i < 8; ++i) v[8 + i] = p[16 + i];
    return v;
}

__device__ __forceinline__ void async_copy16(uint32_t lds_addr, const bf16* gptr) {
    // CDNA5 GLOBAL_LOAD_ASYNC_TO_LDS_B128 (GV mode): per-lane 16B global->LDS DMA
    asm volatile("global_load_async_to_lds_b128 %0, %1, off"
                 :: "v"(lds_addr), "v"((uint64_t)(uintptr_t)gptr) : "memory");
}

__global__ __launch_bounds__(256) void gemm_bf16_wmma(
    const bf16* __restrict__ A, const bf16* __restrict__ W,
    const float* __restrict__ bias, float* __restrict__ Cf, bf16* __restrict__ Cb,
    int M, int N, int K, int act)
{
    __shared__ __attribute__((aligned(16))) bf16 As[2][BM * LDT];
    __shared__ __attribute__((aligned(16))) bf16 Bs[2][BN * LDT];
    int tid  = threadIdx.x;
    int lane = tid & 31, w = tid >> 5;
    int bm = blockIdx.x * BM, bn = blockIdx.y * BN;
    int wm = (w >> 1) * 32, wn = (w & 1) * 32;

    v8f acc[2][2];
#pragma unroll
    for (int i = 0; i < 2; ++i)
#pragma unroll
        for (int j = 0; j < 2; ++j)
#pragma unroll
            for (int e = 0; e < 8; ++e) acc[i][j][e] = 0.0f;

    // issue the 6 wave-wide async DMA ops that fill one 64-wide K slab
    auto issue = [&](int buf, int k0) {
#pragma unroll
        for (int c = 0; c < 4; ++c) {               // A: 128x64 = 1024 16B chunks
            int idx = tid + c * 256;
            int row = idx >> 3, col = (idx & 7) * 8;
            async_copy16((uint32_t)(uintptr_t)&As[buf][row * LDT + col],
                         A + (size_t)(bm + row) * K + k0 + col);
        }
#pragma unroll
        for (int c = 0; c < 2; ++c) {               // B: 64x64 = 512 16B chunks
            int idx = tid + c * 256;
            int row = idx >> 3, col = (idx & 7) * 8;
            async_copy16((uint32_t)(uintptr_t)&Bs[buf][row * LDT + col],
                         W + (size_t)(bn + row) * K + k0 + col);
        }
    };

    const int nk = K / BKS;
    issue(0, 0);
    for (int kt = 0; kt < nk; ++kt) {
        int buf = kt & 1;
        if (kt + 1 < nk) {
            issue(buf ^ 1, (kt + 1) * BKS);          // prefetch next slab into other buffer
            asm volatile("s_wait_asynccnt 6" ::: "memory");  // current slab's 6 ops done
        } else {
            asm volatile("s_wait_asynccnt 0" ::: "memory");
        }
        __syncthreads();                             // all waves' DMA for this slab visible
#pragma unroll
        for (int kk = 0; kk < 2; ++kk) {             // two 16x16x32 k-steps per slab
#pragma unroll
            for (int i = 0; i < 2; ++i) {
                bf16x16 a = frag_a(&As[buf][(wm + i * 16) * LDT + kk * 32], lane);
#pragma unroll
                for (int j = 0; j < 2; ++j) {
                    bf16x16 b = frag_b(&Bs[buf][(wn + j * 16) * LDT + kk * 32], lane);
                    acc[i][j] = __builtin_amdgcn_wmma_f32_16x16x32_bf16(
                        false, a, false, b, (short)0, acc[i][j], false, false);
                }
            }
        }
        __syncthreads();                             // done reading buf; reusable for kt+2
    }

    // epilogue: f32 C/D 16x16: VGPR r -> M = r + 8*(lane>=16), N = lane&15
    int half = lane >> 4, nl = lane & 15;
#pragma unroll
    for (int i = 0; i < 2; ++i)
#pragma unroll
        for (int j = 0; j < 2; ++j) {
            int n = bn + wn + j * 16 + nl;
            float bv = bias ? bias[n] : 0.0f;
#pragma unroll
            for (int r = 0; r < 8; ++r) {
                int m = bm + wm + i * 16 + half * 8 + r;
                if (m >= M || n >= N) continue;
                float v = acc[i][j][r] + bv;
                if (act) v = 0.5f * v * (1.0f + erff(v * 0.70710678118654752f));
                size_t o = (size_t)m * N + n;
                if (Cf) Cf[o] = v;
                if (Cb) Cb[o] = (bf16)v;
            }
        }
}

// ------------------------------------------------- sampled-score sparsity measure
// M[b,h,i] = max_j q_i . k_sample_j  -  (sum_j ...)/L ; one wave per row
__global__ void probsparse_measure(const bf16* __restrict__ Qb, const bf16* __restrict__ Kb,
                                   const int* __restrict__ idxs, float* __restrict__ Mm)
{
    int wave = threadIdx.x >> 5, lane = threadIdx.x & 31;
    int rid = blockIdx.x * 8 + wave;          // rid < B*H*L
    int b = rid >> 14;                        // H*L = 16384
    int h = (rid >> 11) & 7;
    int i = rid & (LSEQ - 1);
    const bf16* q = Qb + ((size_t)(b * LSEQ + i)) * DMOD + h * DK;
    float q0 = (float)q[2 * lane], q1 = (float)q[2 * lane + 1];
    float smax = -3.4e38f, ssum = 0.0f;
    for (int j = 0; j < SK; ++j) {
        int kk = idxs[i * SK + j];
        const bf16* kp = Kb + ((size_t)(b * LSEQ + kk)) * DMOD + h * DK;
        float part = q0 * (float)kp[2 * lane] + q1 * (float)kp[2 * lane + 1];
#pragma unroll
        for (int off = 16; off; off >>= 1) part += __shfl_xor(part, off, 32);
        smax = fmaxf(smax, part);
        ssum += part;
    }
    if (lane == 0) Mm[rid] = smax - ssum * (1.0f / LSEQ);
}

// ---------------------------------------------------------------- top-40 per (b,h)
__global__ void topk40(const float* __restrict__ Mm, int* __restrict__ topidx)
{
    __shared__ float vals[LSEQ];
    __shared__ float rv[256];
    __shared__ int   ri[256];
    int bh = blockIdx.x, tid = threadIdx.x;
    for (int i = tid; i < LSEQ; i += 256) vals[i] = Mm[(size_t)bh * LSEQ + i];
    __syncthreads();
    for (int t = 0; t < NTOP; ++t) {
        float bv = -3.4e38f; int bi = LSEQ;
        for (int i = tid; i < LSEQ; i += 256) {
            float v = vals[i];
            if (v > bv || (v == bv && i < bi)) { bv = v; bi = i; }
        }
        rv[tid] = bv; ri[tid] = bi;
        __syncthreads();
        for (int s = 128; s > 0; s >>= 1) {
            if (tid < s) {
                if (rv[tid + s] > rv[tid] || (rv[tid + s] == rv[tid] && ri[tid + s] < ri[tid])) {
                    rv[tid] = rv[tid + s]; ri[tid] = ri[tid + s];
                }
            }
            __syncthreads();
        }
        if (tid == 0) { topidx[bh * NTOP + t] = ri[0]; vals[ri[0]] = -3.4e38f; }
        __syncthreads();
    }
}

// ---------------------------------------------------------------- vmean[b,h,d]
__global__ void vmean_kernel(const bf16* __restrict__ Vb, float* __restrict__ vmean)
{
    __shared__ float part[256];
    int bh = blockIdx.x, tid = threadIdx.x;
    int b = bh >> 3, h = bh & 7;
    int d = tid & 63, c = tid >> 6;
    float s = 0.0f;
    for (int l = c * 512; l < (c + 1) * 512; ++l)
        s += (float)Vb[((size_t)(b * LSEQ + l)) * DMOD + h * DK + d];
    part[tid] = s;
    __syncthreads();
    if (c == 0)
        vmean[bh * DK + d] = (part[d] + part[64 + d] + part[128 + d] + part[192 + d]) * (1.0f / LSEQ);
}

// -------------------------------------------- ctx default: uniform attention = mean(V)
__global__ void fill_ctx(const float* __restrict__ vmean, bf16* __restrict__ ctxb)
{
    size_t n = (size_t)blockIdx.x * 256 + threadIdx.x;   // < B*L*D
    int c = (int)(n & (DMOD - 1));
    size_t bl = n >> 9;
    int b = (int)(bl >> 11);
    int h = c >> 6, d = c & 63;
    ctxb[n] = (bf16)vmean[((b << 3) + h) * DK + d];
}

// ---------------------------------- full softmax attention for the 40 selected rows
__global__ void top_rows_attn(const bf16* __restrict__ Qb, const bf16* __restrict__ Kb,
                              const bf16* __restrict__ Vb, const int* __restrict__ topidx,
                              bf16* __restrict__ ctxb)
{
    __shared__ float logits[LSEQ];
    __shared__ float red[256];
    __shared__ float qsh[DK];
    int g = blockIdx.x, tid = threadIdx.x;
    int b = g / (NH * NTOP);
    int rem = g % (NH * NTOP);
    int h = rem / NTOP, t = rem % NTOP;
    int l = topidx[(b * NH + h) * NTOP + t];
    size_t qoff = ((size_t)(b * LSEQ + l)) * DMOD + h * DK;
    if (tid < DK) qsh[tid] = (float)Qb[qoff + tid];
    __syncthreads();
    float lmax = -3.4e38f;
    for (int k = tid; k < LSEQ; k += 256) {
        const bf16* kp = Kb + ((size_t)(b * LSEQ + k)) * DMOD + h * DK;
        float s = 0.0f;
#pragma unroll 8
        for (int d2 = 0; d2 < DK; ++d2) s += qsh[d2] * (float)kp[d2];
        logits[k] = s;
        lmax = fmaxf(lmax, s);
    }
    red[tid] = lmax; __syncthreads();
    for (int s = 128; s > 0; s >>= 1) { if (tid < s) red[tid] = fmaxf(red[tid], red[tid + s]); __syncthreads(); }
    float mx = red[0];
    __syncthreads();
    float lsum = 0.0f;
    for (int k = tid; k < LSEQ; k += 256) { float p = __expf(logits[k] - mx); logits[k] = p; lsum += p; }
    red[tid] = lsum; __syncthreads();
    for (int s = 128; s > 0; s >>= 1) { if (tid < s) red[tid] += red[tid + s]; __syncthreads(); }
    float denom = red[0];
    __syncthreads();
    int d = tid & 63, c = tid >> 6;
    float acc = 0.0f;
    for (int k = c * 512; k < (c + 1) * 512; ++k)
        acc += logits[k] * (float)Vb[((size_t)(b * LSEQ + k)) * DMOD + h * DK + d];
    red[tid] = acc; __syncthreads();
    if (c == 0) {
        float tot = red[d] + red[64 + d] + red[128 + d] + red[192 + d];
        ctxb[qoff + d] = (bf16)(tot / denom);
    }
}

// ---------------------------------------------------------------- add + layernorm
__global__ void add_ln(const float* __restrict__ a, const float* __restrict__ b,
                       const float* __restrict__ g, const float* __restrict__ be,
                       float* __restrict__ outf, bf16* __restrict__ outb)
{
    __shared__ float s1[256], s2[256];
    int row = blockIdx.x, tid = threadIdx.x;
    size_t base = (size_t)row * DMOD;
    float v0 = a[base + tid] + b[base + tid];
    float v1 = a[base + tid + 256] + b[base + tid + 256];
    s1[tid] = v0 + v1;
    s2[tid] = v0 * v0 + v1 * v1;
    __syncthreads();
    for (int s = 128; s > 0; s >>= 1) {
        if (tid < s) { s1[tid] += s1[tid + s]; s2[tid] += s2[tid + s]; }
        __syncthreads();
    }
    float mu  = s1[0] * (1.0f / DMOD);
    float var = s2[0] * (1.0f / DMOD) - mu * mu;
    float rstd = rsqrtf(var + 1e-5f);
    float y0 = (v0 - mu) * rstd * g[tid] + be[tid];
    float y1 = (v1 - mu) * rstd * g[tid + 256] + be[tid + 256];
    outf[base + tid]       = y0;
    outf[base + tid + 256] = y1;
    if (outb) { outb[base + tid] = (bf16)y0; outb[base + tid + 256] = (bf16)y1; }
}

// ================================================================= launch
extern "C" void kernel_launch(void* const* d_in, const int* in_sizes, int n_in,
                              void* d_out, int out_size, void* d_ws, size_t ws_size,
                              hipStream_t stream)
{
    const float* x   = (const float*)d_in[0];
    const int*   idxs= (const int*)  d_in[1];
    const float* Wq  = (const float*)d_in[2];
    const float* bq  = (const float*)d_in[3];
    const float* Wk  = (const float*)d_in[4];
    const float* bk  = (const float*)d_in[5];
    const float* Wv  = (const float*)d_in[6];
    const float* bv  = (const float*)d_in[7];
    const float* Wo  = (const float*)d_in[8];
    const float* bo  = (const float*)d_in[9];
    const float* W1  = (const float*)d_in[10];
    const float* b1  = (const float*)d_in[11];
    const float* W2  = (const float*)d_in[12];
    const float* b2  = (const float*)d_in[13];
    const float* g1  = (const float*)d_in[14];
    const float* be1 = (const float*)d_in[15];
    const float* g2  = (const float*)d_in[16];
    const float* be2 = (const float*)d_in[17];
    float* out = (float*)d_out;

    const size_t BLD = (size_t)BDIM * LSEQ * DMOD;   // 4,194,304
    const size_t BLF = (size_t)BDIM * LSEQ * FF;     // 16,777,216
    char* ws = (char*)d_ws;
    size_t off = 0;
    auto alloc = [&](size_t bytes) -> void* {
        void* p = ws + off;
        off = (off + bytes + 255) & ~(size_t)255;
        return p;
    };
    bf16*  xb    = (bf16*) alloc(BLD * 2);
    bf16*  Wqb   = (bf16*) alloc((size_t)DMOD * DMOD * 2);
    bf16*  Wkb   = (bf16*) alloc((size_t)DMOD * DMOD * 2);
    bf16*  Wvb   = (bf16*) alloc((size_t)DMOD * DMOD * 2);
    bf16*  Wob   = (bf16*) alloc((size_t)DMOD * DMOD * 2);
    bf16*  W1b   = (bf16*) alloc((size_t)FF * DMOD * 2);
    bf16*  W2b   = (bf16*) alloc((size_t)DMOD * FF * 2);
    bf16*  Qb    = (bf16*) alloc(BLD * 2);
    bf16*  Kb    = (bf16*) alloc(BLD * 2);
    bf16*  Vb    = (bf16*) alloc(BLD * 2);
    float* Mm    = (float*)alloc((size_t)BDIM * NH * LSEQ * 4);
    int*   tidx  = (int*)  alloc((size_t)BDIM * NH * NTOP * 4);
    float* vmean = (float*)alloc((size_t)BDIM * NH * DK * 4);
    bf16*  ctxb  = (bf16*) alloc(BLD * 2);
    float* attn  = (float*)alloc(BLD * 4);
    float* h1    = (float*)alloc(BLD * 4);
    bf16*  h1b   = (bf16*) alloc(BLD * 2);
    bf16*  gel   = (bf16*) alloc(BLF * 2);
    float* ffo   = (float*)alloc(BLD * 4);
    (void)ws_size; (void)in_sizes; (void)n_in; (void)out_size;

    // f32 -> bf16 conversions
    cvt_f32_bf16<<<(int)(BLD / 256), 256, 0, stream>>>(x, xb, (int)BLD);
    cvt_f32_bf16<<<DMOD * DMOD / 256, 256, 0, stream>>>(Wq, Wqb, DMOD * DMOD);
    cvt_f32_bf16<<<DMOD * DMOD / 256, 256, 0, stream>>>(Wk, Wkb, DMOD * DMOD);
    cvt_f32_bf16<<<DMOD * DMOD / 256, 256, 0, stream>>>(Wv, Wvb, DMOD * DMOD);
    cvt_f32_bf16<<<DMOD * DMOD / 256, 256, 0, stream>>>(Wo, Wob, DMOD * DMOD);
    cvt_f32_bf16<<<FF * DMOD / 256, 256, 0, stream>>>(W1, W1b, FF * DMOD);
    cvt_f32_bf16<<<FF * DMOD / 256, 256, 0, stream>>>(W2, W2b, FF * DMOD);

    const int Mrows = BDIM * LSEQ;   // 8192
    dim3 blk(256);
    // Q, K, V projections (bf16 out for attention)
    gemm_bf16_wmma<<<dim3(Mrows / BM, DMOD / BN), blk, 0, stream>>>(xb, Wqb, bq, nullptr, Qb, Mrows, DMOD, DMOD, 0);
    gemm_bf16_wmma<<<dim3(Mrows / BM, DMOD / BN), blk, 0, stream>>>(xb, Wkb, bk, nullptr, Kb, Mrows, DMOD, DMOD, 0);
    gemm_bf16_wmma<<<dim3(Mrows / BM, DMOD / BN), blk, 0, stream>>>(xb, Wvb, bv, nullptr, Vb, Mrows, DMOD, DMOD, 0);

    // ProbSparse attention
    probsparse_measure<<<BDIM * NH * LSEQ / 8, 256, 0, stream>>>(Qb, Kb, idxs, Mm);
    topk40<<<BDIM * NH, 256, 0, stream>>>(Mm, tidx);
    vmean_kernel<<<BDIM * NH, 256, 0, stream>>>(Vb, vmean);
    fill_ctx<<<(int)(BLD / 256), 256, 0, stream>>>(vmean, ctxb);
    top_rows_attn<<<BDIM * NH * NTOP, 256, 0, stream>>>(Qb, Kb, Vb, tidx, ctxb);

    // output projection + first add&LN
    gemm_bf16_wmma<<<dim3(Mrows / BM, DMOD / BN), blk, 0, stream>>>(ctxb, Wob, bo, attn, nullptr, Mrows, DMOD, DMOD, 0);
    add_ln<<<Mrows, 256, 0, stream>>>(x, attn, g1, be1, h1, h1b);

    // FFN: gelu(h1 @ W1^T + b1) @ W2^T + b2, then final add&LN -> d_out
    gemm_bf16_wmma<<<dim3(Mrows / BM, FF / BN), blk, 0, stream>>>(h1b, W1b, b1, nullptr, gel, Mrows, FF, DMOD, 1);
    gemm_bf16_wmma<<<dim3(Mrows / BM, DMOD / BN), blk, 0, stream>>>(gel, W2b, b2, ffo, nullptr, Mrows, DMOD, FF, 0);
    add_ln<<<Mrows, 256, 0, stream>>>(h1, ffo, g2, be2, out, nullptr);
}
